// AttentionWithBias_54322746360187
// MI455X (gfx1250) — matrix operand order repaired and checked by reading, not compiled, not run on hardware
//
#include <hip/hip_runtime.h>
#include <hip/hip_bf16.h>

#define B_ 2
#define H_ 16
#define S_ 2048
#define D_ 64
#define RS 2052   // padded LDS row stride (floats): stride%64 == 4 -> conflict-free

typedef __attribute__((ext_vector_type(16))) __bf16 v16bf;
typedef __attribute__((ext_vector_type(4)))  __bf16 v4bf;
typedef __attribute__((ext_vector_type(8)))  float  v8f;
typedef __attribute__((ext_vector_type(4)))  float  v4f;

static __device__ __forceinline__ v8f wmma_bf16(v16bf a, v16bf b, v8f c) {
  // 8 args: (neg_a, A, neg_b, B, c_mod, C, reuse_a, reuse_b)
  return __builtin_amdgcn_wmma_f32_16x16x32_bf16(false, a, false, b, (short)0, c,
                                                 false, false);
}

static __device__ __forceinline__ __bf16 bfhi(float f) { return (__bf16)f; }
static __device__ __forceinline__ __bf16 bflo(float f, __bf16 hi) {
  return (__bf16)(f - (float)hi);
}

// ---------------- prep 1: elementwise fp32 -> bf16 hi/lo (K, row-major) ------
__global__ __launch_bounds__(256)
void conv_hilo(const float* __restrict__ src, __bf16* __restrict__ hi,
               __bf16* __restrict__ lo) {
  const size_t i4 = ((size_t)blockIdx.x * 256 + threadIdx.x) * 4;
  v4f f = *(const v4f*)(src + i4);
  v4bf h, l;
  #pragma unroll
  for (int j = 0; j < 4; ++j) {
    __bf16 hj = bfhi(f[j]);
    h[j] = hj;
    l[j] = bflo(f[j], hj);
  }
  *(v4bf*)(hi + i4) = h;
  *(v4bf*)(lo + i4) = l;
}

// ---------------- prep 2: V -> transposed bf16 hi/lo VT[bh][d][k] ------------
__global__ __launch_bounds__(256)
void conv_vt(const float* __restrict__ V, __bf16* __restrict__ vthi,
             __bf16* __restrict__ vtlo) {
  __shared__ __bf16 thi[64][65];
  __shared__ __bf16 tlo[64][65];
  const int tid = threadIdx.x;
  const int bh  = blockIdx.y;
  const int k0  = blockIdx.x * 64;
  const float* Vbh = V + (size_t)bh * S_ * D_;
  #pragma unroll
  for (int it = 0; it < 16; ++it) {
    int idx = tid + it * 256;
    int r = idx >> 6, d = idx & 63;                 // coalesced read V[k0+r][d]
    float f = Vbh[(size_t)(k0 + r) * D_ + d];
    __bf16 h = bfhi(f);
    thi[d][r] = h;
    tlo[d][r] = bflo(f, h);
  }
  __syncthreads();
  #pragma unroll
  for (int it = 0; it < 16; ++it) {
    int idx = tid + it * 256;
    int d = idx >> 6, r = idx & 63;                 // coalesced write VT[d][k0+r]
    size_t o = (size_t)bh * D_ * S_ + (size_t)d * S_ + k0 + r;
    vthi[o] = thi[d][r];
    vtlo[o] = tlo[d][r];
  }
}

// ---------------- main fused attention kernel --------------------------------
template <bool PRE>
__global__ __launch_bounds__(256)
void attn_bias_kernel(const float* __restrict__ Q, const float* __restrict__ K,
                      const float* __restrict__ V, const float* __restrict__ Bias,
                      const unsigned char* __restrict__ Mask,
                      float* __restrict__ Out, float* __restrict__ P,
                      const __bf16* __restrict__ KHi, const __bf16* __restrict__ KLo,
                      const __bf16* __restrict__ VTHi, const __bf16* __restrict__ VTLo)
{
  __shared__ float smem[16 * RS];          // ~128 KB: scores -> probs -> partials
  __shared__ unsigned char mrow_s[16];

  const int tid  = threadIdx.x;
  const int wave = tid >> 5;
  const int lane = tid & 31;
  const int half = lane >> 4;              // 0: lanes 0-15, 1: lanes 16-31
  const int ln   = lane & 15;

  const int bh = blockIdx.x >> 7;          // 128 q-tiles per (b,h)
  const int qt = blockIdx.x & 127;
  const int q0 = qt * 16;
  const int b  = bh >> 4;                  // H_ == 16

  const float* Qbh = Q + (size_t)bh * S_ * D_;
  const float* Kbh = K + (size_t)bh * S_ * D_;
  const float* Vbh = V + (size_t)bh * S_ * D_;
  const float* Bbh = Bias + (size_t)bh * S_ * S_;
  const unsigned char* Mb = Mask + (size_t)b * S_;

  if (tid < 16) mrow_s[tid] = Mb[q0 + tid];
  __syncthreads();

  unsigned char mq[8];
  #pragma unroll
  for (int r = 0; r < 8; ++r) mq[r] = mrow_s[r + half * 8];

  // ---- A operand: Q tile (16x64) as bf16 hi/lo, WMMA A layout ----
  // lane element j = K(kc*32 + half*16 + j); element 8+j = +8; row = ln
  v16bf aHi[2], aLo[2];
  {
    const float* qrow = Qbh + (size_t)(q0 + ln) * D_;
    #pragma unroll
    for (int kc = 0; kc < 2; ++kc) {
      const int kb = kc * 32 + half * 16;
      #pragma unroll
      for (int j = 0; j < 8; ++j) {
        float f0 = qrow[kb + j];
        float f1 = qrow[kb + 8 + j];
        __bf16 h0 = bfhi(f0), h1 = bfhi(f1);
        aHi[kc][j]     = h0;  aLo[kc][j]     = bflo(f0, h0);
        aHi[kc][8 + j] = h1;  aLo[kc][8 + j] = bflo(f1, h1);
      }
    }
  }

  // ---- Phase 1: scores = (Q K^T + bias)/8, masked, into LDS ----
  for (int i = 0; i < 16; ++i) {
    const int ct      = wave * 16 + i;
    const int colbase = ct * 16;

    // B operand: B[k][n] = Key[colbase+n][k]; lane n=ln holds K = kc*32+half*16+j
    v16bf bHi[2], bLo[2];
    if (PRE) {
      const __bf16* khi = KHi + (size_t)bh * S_ * D_ + (size_t)(colbase + ln) * D_;
      const __bf16* klo = KLo + (size_t)bh * S_ * D_ + (size_t)(colbase + ln) * D_;
      #pragma unroll
      for (int kc = 0; kc < 2; ++kc) {
        const int kb = kc * 32 + half * 16;
        bHi[kc] = *(const v16bf*)(khi + kb);
        bLo[kc] = *(const v16bf*)(klo + kb);
      }
    } else {
      const float* krow = Kbh + (size_t)(colbase + ln) * D_;
      #pragma unroll
      for (int kc = 0; kc < 2; ++kc) {
        const int kb = kc * 32 + half * 16;
        #pragma unroll
        for (int j = 0; j < 16; ++j) {
          float f = krow[kb + j];
          __bf16 h = bfhi(f);
          bHi[kc][j] = h;  bLo[kc][j] = bflo(f, h);
        }
      }
    }

    v8f acc = {};
    #pragma unroll
    for (int kc = 0; kc < 2; ++kc) {
      acc = wmma_bf16(aHi[kc], bHi[kc], acc);
      acc = wmma_bf16(aHi[kc], bLo[kc], acc);
      acc = wmma_bf16(aLo[kc], bHi[kc], acc);
    }

    const bool mc = Mb[colbase + ln] != 0;
    const float* brow = Bbh + (size_t)q0 * S_ + colbase + ln;
    #pragma unroll
    for (int r = 0; r < 8; ++r) {
      const int row = r + half * 8;        // C layout: M = r + 8*(lane/16), N = ln
      float bv = __builtin_nontemporal_load(brow + (size_t)row * S_);
      bool valid = mc && (mq[r] != 0);
      float sc = valid ? (acc[r] + bv) * 0.125f : -1.0e9f;
      smem[row * RS + colbase + ln] = sc;
    }
  }
  __syncthreads();

  // ---- Phase 2: row softmax in LDS, stream p_attn to HBM ----
  {
    const int row = tid >> 4;
    const int t16 = tid & 15;
    float* srow = &smem[row * RS];

    float mx = -3.0e38f;
    for (int i = t16; i < S_; i += 16) mx = fmaxf(mx, srow[i]);
    #pragma unroll
    for (int m = 8; m >= 1; m >>= 1) mx = fmaxf(mx, __shfl_xor(mx, m, 16));

    float sum = 0.f;
    for (int i = t16; i < S_; i += 16) {
      float e = __expf(srow[i] - mx);
      srow[i] = e;
      sum += e;
    }
    #pragma unroll
    for (int m = 8; m >= 1; m >>= 1) sum += __shfl_xor(sum, m, 16);
    const float inv = 1.0f / sum;

    float* prow = P + (size_t)bh * S_ * S_ + (size_t)(q0 + row) * S_;
    for (int i = t16; i < S_ / 4; i += 16) {
      v4f e4 = *(const v4f*)(srow + i * 4);
      e4 *= inv;
      *(v4f*)(srow + i * 4) = e4;                          // keep P in LDS
      __builtin_nontemporal_store(e4, (v4f*)(prow + (size_t)i * 4));
    }
  }
  __syncthreads();

  // ---- Phase 3: out = P @ V, K split across the 8 waves ----
  {
    const int kw = wave * 256;
    v8f zero = {};
    v8f acc2[4];
    #pragma unroll
    for (int nt = 0; nt < 4; ++nt) acc2[nt] = zero;

    for (int kc8 = 0; kc8 < 8; ++kc8) {
      const int kb = kw + kc8 * 32;

      // A operand: P chunk from LDS (bf16 A layout), padded stride -> no conflicts
      v16bf pHi, pLo;
      {
        const float* pr = &smem[ln * RS + kb + half * 16];
        #pragma unroll
        for (int j = 0; j < 8; ++j) {
          float f0 = pr[j];
          float f1 = pr[8 + j];
          __bf16 h0 = bfhi(f0), h1 = bfhi(f1);
          pHi[j]     = h0;  pLo[j]     = bflo(f0, h0);
          pHi[8 + j] = h1;  pLo[8 + j] = bflo(f1, h1);
        }
      }

      #pragma unroll
      for (int nt = 0; nt < 4; ++nt) {
        const int dbase = nt * 16;
        // B operand: element j = V[kb+half*16+j][dbase+ln] = VT[dbase+ln][kb+half*16+j]
        v16bf vHi, vLo;
        if (PRE) {
          const size_t ro = (size_t)bh * D_ * S_ + (size_t)(dbase + ln) * S_;
          vHi = *(const v16bf*)(VTHi + ro + kb + half * 16);
          vLo = *(const v16bf*)(VTLo + ro + kb + half * 16);
        } else {
          const float* vp = Vbh + (size_t)(kb + half * 16) * D_ + dbase + ln;
          #pragma unroll
          for (int j = 0; j < 16; ++j) {
            float f = vp[(size_t)j * D_];
            __bf16 h = bfhi(f);
            vHi[j] = h;  vLo[j] = bflo(f, h);
          }
        }
        acc2[nt] = wmma_bf16(pHi, vHi, acc2[nt]);
        acc2[nt] = wmma_bf16(pHi, vLo, acc2[nt]);
        acc2[nt] = wmma_bf16(pLo, vHi, acc2[nt]);
      }
    }

    __syncthreads();                       // all LDS P reads complete
    float* part = smem;                    // reuse front of smem for partials
    #pragma unroll
    for (int nt = 0; nt < 4; ++nt)
      #pragma unroll
      for (int r = 0; r < 8; ++r)
        part[((wave * 4 + nt) * 8 + r) * 32 + lane] = acc2[nt][r];
    __syncthreads();

    for (int idx = tid; idx < 1024; idx += 256) {
      const int nt  = idx >> 8;
      const int rem = idx & 255;
      const int r   = rem >> 5;
      const int l2  = rem & 31;
      float s = 0.f;
      #pragma unroll
      for (int w = 0; w < 8; ++w)
        s += part[((w * 4 + nt) * 8 + r) * 32 + l2];
      const int mrow = r + ((l2 >> 4) << 3);
      const int ncol = l2 & 15;
      Out[(size_t)bh * S_ * D_ + (size_t)(q0 + mrow) * D_ + nt * 16 + ncol] = s;
    }
  }
}

extern "C" void kernel_launch(void* const* d_in, const int* in_sizes, int n_in,
                              void* d_out, int out_size, void* d_ws, size_t ws_size,
                              hipStream_t stream) {
  (void)in_sizes; (void)n_in; (void)out_size;
  const float* q    = (const float*)d_in[0];
  const float* k    = (const float*)d_in[1];
  const float* v    = (const float*)d_in[2];
  const float* bias = (const float*)d_in[3];
  const unsigned char* mask = (const unsigned char*)d_in[4];

  float* out = (float*)d_out;
  float* p   = out + (size_t)B_ * H_ * S_ * D_;   // tuple: (out, p_attn) flat

  const size_t N    = (size_t)B_ * H_ * S_ * D_;  // 4M elements
  const size_t need = 4 * N * sizeof(__bf16);     // KHi,KLo,VTHi,VTLo = 32 MB
  const dim3 grid(B_ * H_ * (S_ / 16));           // 4096 blocks

  if (ws_size >= need) {
    __bf16* khi  = (__bf16*)d_ws;
    __bf16* klo  = khi + N;
    __bf16* vthi = klo + N;
    __bf16* vtlo = vthi + N;
    conv_hilo<<<dim3((unsigned)(N / 1024)), 256, 0, stream>>>(k, khi, klo);
    conv_vt<<<dim3(S_ / 64, B_ * H_), 256, 0, stream>>>(v, vthi, vtlo);
    attn_bias_kernel<true><<<grid, 256, 0, stream>>>(q, k, v, bias, mask, out, p,
                                                     khi, klo, vthi, vtlo);
  } else {
    attn_bias_kernel<false><<<grid, 256, 0, stream>>>(q, k, v, bias, mask, out, p,
                                                      nullptr, nullptr, nullptr,
                                                      nullptr);
  }
}
